// MViT_73796128080562
// MI455X (gfx1250) — matrix operand order repaired and checked by reading, not compiled, not run on hardware
//
#include <hip/hip_runtime.h>
#include <math.h>

typedef __attribute__((ext_vector_type(16))) _Float16 v16h;
typedef __attribute__((ext_vector_type(8)))  _Float16 h8;
typedef __attribute__((ext_vector_type(8)))  float    v8f;

#define TB 256

__device__ __forceinline__ float gelu_exact(float v){
  return 0.5f * v * (1.0f + erff(v * 0.70710678118654752f));
}

__device__ __forceinline__ v16h cat16(h8 lo, h8 hi){
  return __builtin_shufflevector(lo, hi, 0,1,2,3,4,5,6,7,8,9,10,11,12,13,14,15);
}

// ---------------- patch embed: 7x7 s4 p3 conv (NCHW in, tokens out) ----------
__global__ void k_patch_embed(const float* __restrict__ x, const float* __restrict__ pw,
                              const float* __restrict__ pb, const float* __restrict__ pos,
                              float* __restrict__ tok, int B){
  int id = blockIdx.x*blockDim.x + threadIdx.x;
  int total = B*56*56*96;
  if (id >= total) return;
  int oc = id % 96; int t = id / 96;
  int ox = t % 56; t /= 56; int oy = t % 56; int b = t / 56;
  float acc = pb[oc];
  for (int c = 0; c < 3; ++c)
    for (int ky = 0; ky < 7; ++ky){
      int iy = oy*4 - 3 + ky; if (iy < 0 || iy >= 224) continue;
      for (int kx = 0; kx < 7; ++kx){
        int ix = ox*4 - 3 + kx; if (ix < 0 || ix >= 224) continue;
        acc += x[((size_t)(b*3 + c)*224 + iy)*224 + ix] * pw[((ky*7 + kx)*3 + c)*96 + oc];
      }
    }
  int tkn = 1 + oy*56 + ox;
  tok[((size_t)b*3137 + tkn)*96 + oc] = acc + pos[(size_t)tkn*96 + oc];
}

__global__ void k_cls_token(const float* __restrict__ cls, const float* __restrict__ pos,
                            float* __restrict__ tok, int B){
  int id = blockIdx.x*blockDim.x + threadIdx.x;
  if (id >= B*96) return;
  int c = id % 96, b = id / 96;
  tok[(size_t)b*3137*96 + c] = cls[c] + pos[c];
}

// ---------------- generic LayerNorm over last dim ---------------------------
__global__ void __launch_bounds__(TB)
k_layernorm(const float* __restrict__ in, const float* __restrict__ g,
            const float* __restrict__ bt, float* __restrict__ out, int C){
  __shared__ float s1[TB], s2[TB];
  size_t row = blockIdx.x;
  const float* p = in + row*C;
  float sum = 0.f, sq = 0.f;
  for (int c = threadIdx.x; c < C; c += TB){ float v = p[c]; sum += v; sq += v*v; }
  s1[threadIdx.x] = sum; s2[threadIdx.x] = sq; __syncthreads();
  for (int o = TB/2; o > 0; o >>= 1){
    if (threadIdx.x < o){ s1[threadIdx.x]+=s1[threadIdx.x+o]; s2[threadIdx.x]+=s2[threadIdx.x+o]; }
    __syncthreads();
  }
  float mu  = s1[0]/C;
  float var = s2[0]/C - mu*mu;
  float rs  = rsqrtf(var + 1e-6f);
  for (int c = threadIdx.x; c < C; c += TB)
    out[row*C + c] = (p[c]-mu)*rs*g[c] + bt[c];
}

// ---------------- WMMA GEMM: C = act(A@W + bias) [+ resid] ------------------
// A:[M,K] f32, W:[K,N] f32; f16 WMMA compute, f32 accumulate.
// REQUIRES: K % 32 == 0, N % 32 == 0 (true for every GEMM in this model).
// 256 threads = 8 waves (4x2 grid); block tile 128(M) x 64(N); K step 32.
// Each wave owns a 32x32 tile: 4 wmma per K-step with A/B fragment reuse.
// Double-buffered LDS + register pipeline: one barrier per K-step, the next
// K-slab's global loads are in flight while the current slab's WMMAs run.
__global__ void __launch_bounds__(256)
k_gemm_wmma(const float* __restrict__ A, const float* __restrict__ Wt,
            const float* __restrict__ bias, const float* __restrict__ resid,
            float* __restrict__ C, int M, int N, int K, int act){
  // stride 40 halves (80B) keeps every fragment offset 16B-aligned
  __shared__ _Float16 sA [2][128][40];  // [buf][m][k]
  __shared__ _Float16 sBT[2][64][40];   // [buf][n][k]  (transposed during staging)
  const int tid  = threadIdx.x;
  const int lane = tid & 31, wid = tid >> 5;
  const int wr = wid >> 1, wn = wid & 1;           // wave grid 4 x 2
  const int m0 = blockIdx.y * 128, n0 = blockIdx.x * 64;

  // staging coordinates (branch-free)
  const int ar = tid >> 1;             // 0..127  A row
  const int ac = (tid & 1) * 16;       // 0,16    A k-offset (16 consecutive k)
  const int bk = tid >> 3;             // 0..31   B k row
  const int bc = (tid & 7) * 8;        // 0..56   B n-offset
  const int arow_g = (m0 + ar < M) ? (m0 + ar) : (M - 1);      // clamp
  const int bcol_g = (n0 + bc + 8 <= N) ? (n0 + bc) : (N - 8); // clamp (N%8==0)
  const float* aptr = A + (size_t)arow_g*K + ac;

  float4 fa0, fa1, fa2, fa3, fb0, fb1;
  auto loadRegs = [&](int k0){
    const float4* ap = reinterpret_cast<const float4*>(aptr + k0);
    fa0 = ap[0]; fa1 = ap[1]; fa2 = ap[2]; fa3 = ap[3];
    const float4* bp = reinterpret_cast<const float4*>(Wt + (size_t)(k0 + bk)*N + bcol_g);
    fb0 = bp[0]; fb1 = bp[1];
  };
  auto storeLds = [&](int buf){
    h8 h0, h1;
    h0[0]=(_Float16)fa0.x; h0[1]=(_Float16)fa0.y; h0[2]=(_Float16)fa0.z; h0[3]=(_Float16)fa0.w;
    h0[4]=(_Float16)fa1.x; h0[5]=(_Float16)fa1.y; h0[6]=(_Float16)fa1.z; h0[7]=(_Float16)fa1.w;
    h1[0]=(_Float16)fa2.x; h1[1]=(_Float16)fa2.y; h1[2]=(_Float16)fa2.z; h1[3]=(_Float16)fa2.w;
    h1[4]=(_Float16)fa3.x; h1[5]=(_Float16)fa3.y; h1[6]=(_Float16)fa3.z; h1[7]=(_Float16)fa3.w;
    *reinterpret_cast<h8*>(&sA[buf][ar][ac])     = h0;
    *reinterpret_cast<h8*>(&sA[buf][ar][ac + 8]) = h1;
    sBT[buf][bc+0][bk] = (_Float16)fb0.x; sBT[buf][bc+1][bk] = (_Float16)fb0.y;
    sBT[buf][bc+2][bk] = (_Float16)fb0.z; sBT[buf][bc+3][bk] = (_Float16)fb0.w;
    sBT[buf][bc+4][bk] = (_Float16)fb1.x; sBT[buf][bc+5][bk] = (_Float16)fb1.y;
    sBT[buf][bc+6][bk] = (_Float16)fb1.z; sBT[buf][bc+7][bk] = (_Float16)fb1.w;
  };

  v8f acc00 = {}, acc01 = {}, acc10 = {}, acc11 = {};
  loadRegs(0);
  storeLds(0);
  int cur = 0;
  for (int k0 = 0; k0 < K; k0 += 32){
    __syncthreads();                       // LDS buffer `cur` ready
    const bool more = (k0 + 32 < K);
    if (more) loadRegs(k0 + 32);           // next slab in flight during compute
    if (k0 + 64 < K){                      // global_prefetch_b8 two slabs ahead
      __builtin_prefetch(aptr + k0 + 64, 0, 1);
      __builtin_prefetch(Wt + (size_t)(k0 + 64 + bk)*N + bcol_g, 0, 1);
    }
    // ---- fragments (all 16B LDS vector loads) ----
    // A frag: lane<16 -> K 0..7 & 16..23 ; lane>=16 -> K 8..15 & 24..31
    const int kb = (lane >> 4) * 8;
    const int r0 = wr*32 + (lane & 15);
    v16h afr0 = cat16(*reinterpret_cast<const h8*>(&sA[cur][r0     ][kb]),
                      *reinterpret_cast<const h8*>(&sA[cur][r0     ][16 + kb]));
    v16h afr1 = cat16(*reinterpret_cast<const h8*>(&sA[cur][r0 + 16][kb]),
                      *reinterpret_cast<const h8*>(&sA[cur][r0 + 16][16 + kb]));
    // B frag: lane<16 -> K 0..7 & 16..23 ; lane>=16 -> K 8..15 & 24..31
    const int c0  = wn*32 + (lane & 15);
    const int kb2 = (lane >> 4) * 8;
    v16h bfr0 = cat16(*reinterpret_cast<const h8*>(&sBT[cur][c0     ][kb2]),
                      *reinterpret_cast<const h8*>(&sBT[cur][c0     ][16 + kb2]));
    v16h bfr1 = cat16(*reinterpret_cast<const h8*>(&sBT[cur][c0 + 16][kb2]),
                      *reinterpret_cast<const h8*>(&sBT[cur][c0 + 16][16 + kb2]));

    acc00 = __builtin_amdgcn_wmma_f32_16x16x32_f16(false, afr0, false, bfr0, (short)0, acc00, false, false);
    acc01 = __builtin_amdgcn_wmma_f32_16x16x32_f16(false, afr0, false, bfr1, (short)0, acc01, false, false);
    acc10 = __builtin_amdgcn_wmma_f32_16x16x32_f16(false, afr1, false, bfr0, (short)0, acc10, false, false);
    acc11 = __builtin_amdgcn_wmma_f32_16x16x32_f16(false, afr1, false, bfr1, (short)0, acc11, false, false);

    if (more) storeLds(cur ^ 1);           // fill the other buffer (no barrier needed)
    cur ^= 1;
  }

  // ---- store: VGPR e -> row e (lanes 0..15) / e+8 (lanes 16..31), col = lane%16
  const int rbase = m0 + wr*32 + 8*(lane >> 4);
  const int cbase = n0 + wn*32 + (lane & 15);
  v8f accs[4] = {acc00, acc01, acc10, acc11};
#pragma unroll
  for (int s = 0; s < 4; ++s){
    const int rr = rbase + (s >> 1)*16;
    const int cc = cbase + (s & 1)*16;
    if (cc < N){
#pragma unroll
      for (int e = 0; e < 8; ++e){
        int r = rr + e;
        if (r < M){
          float v = accs[s][e] + bias[cc];
          if (act == 1) v = gelu_exact(v);
          if (resid) v += resid[(size_t)r*N + cc];
          C[(size_t)r*N + cc] = v;
        }
      }
    }
  }
}

// ---------------- fused dwconv3x3(stride s)+LN for one of q/k/v -------------
// qkv:[B,Ntok,3*heads*hd], out:[B,heads,1+OH*OW,hd]
__global__ void __launch_bounds__(128)
k_pool_ln(const float* __restrict__ qkv, const float* __restrict__ w,
          const float* __restrict__ g, const float* __restrict__ bt,
          float* __restrict__ out, int Ntok, int H, int W, int heads, int hd,
          int comp, int s, int OH, int OW){
  __shared__ float s1[128], s2[128];
  int Nt = 1 + OH*OW;
  int tokid = blockIdx.x % Nt;
  int bh = blockIdx.x / Nt;
  int h = bh % heads, b = bh / heads;
  int rowStride = 3*heads*hd;
  int chanOff = (comp*heads + h)*hd;
  int c = threadIdx.x;
  float val = 0.f;
  if (c < hd){
    if (tokid == 0){
      val = qkv[(size_t)(b*Ntok)*rowStride + chanOff + c];
    } else {
      int t = tokid - 1, oy = t / OW, ox = t % OW;
      for (int ky = 0; ky < 3; ++ky){
        int iy = oy*s - 1 + ky; if (iy < 0 || iy >= H) continue;
        for (int kx = 0; kx < 3; ++kx){
          int ix = ox*s - 1 + kx; if (ix < 0 || ix >= W) continue;
          val += qkv[(size_t)(b*Ntok + 1 + iy*W + ix)*rowStride + chanOff + c]
               * w[(ky*3 + kx)*hd + c];
        }
      }
    }
  }
  float m = (c < hd) ? val : 0.f;
  s1[threadIdx.x] = m; s2[threadIdx.x] = m*m;
  __syncthreads();
  for (int o = 64; o > 0; o >>= 1){
    if (threadIdx.x < o){ s1[threadIdx.x]+=s1[threadIdx.x+o]; s2[threadIdx.x]+=s2[threadIdx.x+o]; }
    __syncthreads();
  }
  float mu = s1[0]/hd, var = s2[0]/hd - mu*mu;
  float rs = rsqrtf(var + 1e-6f);
  if (c < hd)
    out[((size_t)bh*Nt + tokid)*hd + c] = (val-mu)*rs*g[c] + bt[c];
}

// ---------------- attention scores + rel-pos bias + softmax (one row) -------
__global__ void __launch_bounds__(TB)
k_attn_softmax(const float* __restrict__ q, const float* __restrict__ kmat,
               const float* __restrict__ relh, const float* __restrict__ relw,
               float* __restrict__ attn, int Nq, int Nk, int hd,
               int qh, int qw, int kh, int kw, float scale){
  __shared__ float qv[128];
  __shared__ float rH[32], rW[32];
  __shared__ float red[TB];
  int qi = blockIdx.x % Nq;
  int bh = blockIdx.x / Nq;
  const float* qp = q + ((size_t)bh*Nq + qi)*hd;
  for (int c = threadIdx.x; c < hd; c += TB) qv[c] = qp[c];
  __syncthreads();
  if (qi > 0){
    int qy = (qi-1)/qw, qx = (qi-1)%qw;
    int t = threadIdx.x;
    if (t < 32){
      if (t < kh){
        float qr = fmaxf((float)kh/(float)qh, 1.f);
        float kr = fmaxf((float)qh/(float)kh, 1.f);
        int d = (int)(qy*qr - t*kr + (kh-1)*kr);
        const float* rp = relh + (size_t)d*hd;
        float a = 0.f; for (int c = 0; c < hd; ++c) a += qv[c]*rp[c];
        rH[t] = a;
      }
    } else if (t < 64){
      int tw = t - 32;
      if (tw < kw){
        float qr = fmaxf((float)kw/(float)qw, 1.f);
        float kr = fmaxf((float)qw/(float)kw, 1.f);
        int d = (int)(qx*qr - tw*kr + (kw-1)*kr);
        const float* rp = relw + (size_t)d*hd;
        float a = 0.f; for (int c = 0; c < hd; ++c) a += qv[c]*rp[c];
        rW[tw] = a;
      }
    }
  }
  __syncthreads();
  float* arow = attn + (size_t)blockIdx.x * Nk;
  float lmax = -1e30f;
  for (int ki = threadIdx.x; ki < Nk; ki += TB){
    const float* kp = kmat + ((size_t)bh*Nk + ki)*hd;
    float d = 0.f;
    for (int c = 0; c < hd; ++c) d += qv[c]*kp[c];
    d *= scale;
    if (qi > 0 && ki > 0){
      int ky = (ki-1)/kw, kx = (ki-1)%kw;
      d += rH[ky] + rW[kx];
    }
    arow[ki] = d;
    lmax = fmaxf(lmax, d);
  }
  red[threadIdx.x] = lmax; __syncthreads();
  for (int o = TB/2; o > 0; o >>= 1){
    if (threadIdx.x < o) red[threadIdx.x] = fmaxf(red[threadIdx.x], red[threadIdx.x+o]);
    __syncthreads();
  }
  float m = red[0]; __syncthreads();
  float lsum = 0.f;
  for (int ki = threadIdx.x; ki < Nk; ki += TB){
    float e = __expf(arow[ki] - m);
    arow[ki] = e; lsum += e;
  }
  red[threadIdx.x] = lsum; __syncthreads();
  for (int o = TB/2; o > 0; o >>= 1){
    if (threadIdx.x < o) red[threadIdx.x] += red[threadIdx.x+o];
    __syncthreads();
  }
  float inv = 1.f / red[0];
  for (int ki = threadIdx.x; ki < Nk; ki += TB) arow[ki] *= inv;
}

// ---------------- o = attn@v + q, written as [B,Nq,heads*hd] ----------------
__global__ void __launch_bounds__(128)
k_attn_out(const float* __restrict__ attn, const float* __restrict__ v,
           const float* __restrict__ q, float* __restrict__ o,
           int heads, int Nq, int Nk, int hd){
  int qi = blockIdx.x % Nq;
  int bh = blockIdx.x / Nq;
  int h = bh % heads, b = bh / heads;
  int c = threadIdx.x;
  if (c >= hd) return;
  const float* arow = attn + (size_t)blockIdx.x * Nk;
  float acc = q[((size_t)bh*Nq + qi)*hd + c];
  for (int ki = 0; ki < Nk; ++ki)
    acc += arow[ki] * v[((size_t)bh*Nk + ki)*hd + c];
  int dm = heads*hd;
  o[((size_t)(b*Nq + qi))*dm + h*hd + c] = acc;
}

// ---------------- maxpool skip (k=s+1, pad=1), cls passthrough --------------
__global__ void k_maxpool(const float* __restrict__ in, float* __restrict__ out,
                          int B, int H, int W, int C, int s, int OH, int OW){
  int id = blockIdx.x*blockDim.x + threadIdx.x;
  int Nout = 1 + OH*OW;
  size_t total = (size_t)B*Nout*C;
  if ((size_t)id >= total) return;
  int c = id % C; int t = (id / C) % Nout; int b = id / (C*Nout);
  int Nin = 1 + H*W;
  float v;
  if (t == 0) v = in[((size_t)b*Nin)*C + c];
  else {
    int oy = (t-1)/OW, ox = (t-1)%OW;
    v = -INFINITY;
    int k = s + 1, pad = k/2;
    for (int ky = 0; ky < k; ++ky){
      int iy = oy*s - pad + ky; if (iy < 0 || iy >= H) continue;
      for (int kx = 0; kx < k; ++kx){
        int ix = ox*s - pad + kx; if (ix < 0 || ix >= W) continue;
        v = fmaxf(v, in[((size_t)b*Nin + 1 + iy*W + ix)*C + c]);
      }
    }
  }
  out[((size_t)b*Nout + t)*C + c] = v;
}

// ---------------- final: LN(cls) -> head GEMM -> softmax --------------------
__global__ void __launch_bounds__(TB)
k_head(const float* __restrict__ x, const float* __restrict__ nw, const float* __restrict__ nb,
       const float* __restrict__ hw, const float* __restrict__ hb,
       float* __restrict__ out, int Ntok, int C, int NC){
  __shared__ float nrm[768];
  __shared__ float lg[1000];
  __shared__ float red[TB];
  int b = blockIdx.x;
  const float* row = x + (size_t)b*Ntok*C;   // token 0
  float sum = 0.f, sq = 0.f;
  for (int c = threadIdx.x; c < C; c += TB){ float v = row[c]; sum += v; sq += v*v; }
  red[threadIdx.x] = sum; __syncthreads();
  for (int o = TB/2; o > 0; o >>= 1){ if (threadIdx.x < o) red[threadIdx.x]+=red[threadIdx.x+o]; __syncthreads(); }
  float mu = red[0]/C; __syncthreads();
  red[threadIdx.x] = sq; __syncthreads();
  for (int o = TB/2; o > 0; o >>= 1){ if (threadIdx.x < o) red[threadIdx.x]+=red[threadIdx.x+o]; __syncthreads(); }
  float var = red[0]/C - mu*mu;
  float rs = rsqrtf(var + 1e-6f);
  for (int c = threadIdx.x; c < C; c += TB) nrm[c] = (row[c]-mu)*rs*nw[c] + nb[c];
  __syncthreads();
  for (int j = threadIdx.x; j < NC; j += TB){
    float a = hb[j];
    for (int c = 0; c < C; ++c) a += nrm[c]*hw[(size_t)c*NC + j];
    lg[j] = a;
  }
  __syncthreads();
  float lm = -1e30f;
  for (int j = threadIdx.x; j < NC; j += TB) lm = fmaxf(lm, lg[j]);
  red[threadIdx.x] = lm; __syncthreads();
  for (int o = TB/2; o > 0; o >>= 1){ if (threadIdx.x < o) red[threadIdx.x]=fmaxf(red[threadIdx.x],red[threadIdx.x+o]); __syncthreads(); }
  float m = red[0]; __syncthreads();
  float ls = 0.f;
  for (int j = threadIdx.x; j < NC; j += TB){ float e = __expf(lg[j]-m); lg[j] = e; ls += e; }
  red[threadIdx.x] = ls; __syncthreads();
  for (int o = TB/2; o > 0; o >>= 1){ if (threadIdx.x < o) red[threadIdx.x]+=red[threadIdx.x+o]; __syncthreads(); }
  float inv = 1.f / red[0];
  for (int j = threadIdx.x; j < NC; j += TB) out[(size_t)b*NC + j] = lg[j]*inv;
}

// =============================== host side ==================================
extern "C" void kernel_launch(void* const* d_in, const int* in_sizes, int n_in,
                              void* d_out, int out_size, void* d_ws, size_t ws_size,
                              hipStream_t stream){
  (void)in_sizes; (void)n_in; (void)out_size; (void)ws_size;
  const int B = 8, hd = 96;
  // ---- parameter table (setup_inputs()/init_params() insertion order) ----
  int pi = 0;
  const float* x_in    = (const float*)d_in[pi++];
  const float* patch_w = (const float*)d_in[pi++];
  const float* patch_b = (const float*)d_in[pi++];
  const float* cls_tok = (const float*)d_in[pi++];
  const float* pos     = (const float*)d_in[pi++];
  struct Blk {
    const float *n1w,*n1b,*qkvw,*qkvb,*projw,*projb,*pqw,*pkw,*pvw,
                *nqw,*nqb,*nkw,*nkb,*nvw,*nvb,*relh,*relw,*n2w,*n2b,
                *fc1w,*fc1b,*fc2w,*fc2b,*skw,*skb;
  } blk[16];
  static const int dimA[16]  = {96,96,192,192,384,384,384,384,384,384,384,384,384,384,384,768};
  static const int doutA[16] = {96,192,192,384,384,384,384,384,384,384,384,384,384,384,768,768};
  static const int headsA[16]= {1,2,2,4,4,4,4,4,4,4,4,4,4,4,8,8};
  static const int sqA[16]   = {1,2,1,2,1,1,1,1,1,1,1,1,1,1,2,1};
  static const int skvA[16]  = {4,2,2,1,1,1,1,1,1,1,1,1,1,1,1,1};
  for (int i = 0; i < 16; ++i){
    Blk& k = blk[i];
    k.n1w  =(const float*)d_in[pi++]; k.n1b  =(const float*)d_in[pi++];
    k.qkvw =(const float*)d_in[pi++]; k.qkvb =(const float*)d_in[pi++];
    k.projw=(const float*)d_in[pi++]; k.projb=(const float*)d_in[pi++];
    k.pqw  =(const float*)d_in[pi++]; k.pkw  =(const float*)d_in[pi++];
    k.pvw  =(const float*)d_in[pi++];
    k.nqw  =(const float*)d_in[pi++]; k.nqb  =(const float*)d_in[pi++];
    k.nkw  =(const float*)d_in[pi++]; k.nkb  =(const float*)d_in[pi++];
    k.nvw  =(const float*)d_in[pi++]; k.nvb  =(const float*)d_in[pi++];
    k.relh =(const float*)d_in[pi++]; k.relw =(const float*)d_in[pi++];
    k.n2w  =(const float*)d_in[pi++]; k.n2b  =(const float*)d_in[pi++];
    k.fc1w =(const float*)d_in[pi++]; k.fc1b =(const float*)d_in[pi++];
    k.fc2w =(const float*)d_in[pi++]; k.fc2b =(const float*)d_in[pi++];
    if (dimA[i] != doutA[i]){ k.skw=(const float*)d_in[pi++]; k.skb=(const float*)d_in[pi++]; }
    else { k.skw = nullptr; k.skb = nullptr; }
  }
  const float* norm_w = (const float*)d_in[pi++];
  const float* norm_b = (const float*)d_in[pi++];
  const float* head_w = (const float*)d_in[pi++];
  const float* head_b = (const float*)d_in[pi++];

  // ---- workspace carving (float offsets) ----
  float* ws = (float*)d_ws;
  size_t off = 0;
  auto carve = [&](size_t n){ float* p = ws + off; off += n; return p; };
  float* X0  = carve(2500000);   // token ping
  float* X1  = carve(2500000);   // token pong
  float* XN  = carve(2500000);   // layernorm output
  float* XM  = carve(2500000);   // post-attention x
  float* QKV = carve(15000000);  // qkv projection
  float* QP  = carve(2500000);   // pooled+LN q
  float* KP  = carve(2500000);
  float* VP  = carve(2500000);
  float* ATT = carve(10000000);  // attention scores; reused as fc1 hidden
  float* OB  = carve(2500000);   // attn output (transposed)
  float* SK  = carve(5000000);   // skip projection
  float* SKP = carve(1600000);   // pooled skip

  // ---- patch embed + token assembly ----
  {
    int total = B*56*56*96;
    k_patch_embed<<<(total+255)/256, 256, 0, stream>>>(x_in, patch_w, patch_b, pos, X0, B);
    k_cls_token<<<(B*96+255)/256, 256, 0, stream>>>(cls_tok, pos, X0, B);
  }

  int H = 56, W = 56, N = 3137;
  float* xcur = X0; float* xnext = X1;
  const float scale = 1.0f / sqrtf((float)hd);

  for (int i = 0; i < 16; ++i){
    const Blk& bk = blk[i];
    const int dim = dimA[i], dd = doutA[i], hh = headsA[i], sq = sqA[i], skv = skvA[i];
    const int rows = B*N;

    // xn = LN(x)
    k_layernorm<<<rows, TB, 0, stream>>>(xcur, bk.n1w, bk.n1b, XN, dim);

    // qkv = xn @ qkv_w + qkv_b   (WMMA)
    {
      dim3 g((3*dd+63)/64, (rows+127)/128);
      k_gemm_wmma<<<g, 256, 0, stream>>>(XN, bk.qkvw, bk.qkvb, nullptr, QKV, rows, 3*dd, dim, 0);
    }

    const int OH = (H-1)/sq + 1,  OW = (W-1)/sq + 1;
    const int KH = (H-1)/skv + 1, KW = (W-1)/skv + 1;
    const int NqT = 1 + OH*OW, NkT = 1 + KH*KW;

    // pooled + LN q/k/v
    k_pool_ln<<<B*hh*NqT, 128, 0, stream>>>(QKV, bk.pqw, bk.nqw, bk.nqb, QP, N, H, W, hh, hd, 0, sq,  OH, OW);
    k_pool_ln<<<B*hh*NkT, 128, 0, stream>>>(QKV, bk.pkw, bk.nkw, bk.nkb, KP, N, H, W, hh, hd, 1, skv, KH, KW);
    k_pool_ln<<<B*hh*NkT, 128, 0, stream>>>(QKV, bk.pvw, bk.nvw, bk.nvb, VP, N, H, W, hh, hd, 2, skv, KH, KW);

    // attn = softmax(scale*q@k^T + rel bias)
    k_attn_softmax<<<B*hh*NqT, TB, 0, stream>>>(QP, KP, bk.relh, bk.relw, ATT,
                                                NqT, NkT, hd, OH, OW, KH, KW, scale);
    // o = attn@v + q  -> [B, NqT, dd]
    k_attn_out<<<B*hh*NqT, 128, 0, stream>>>(ATT, VP, QP, OB, hh, NqT, NkT, hd);

    // skip path
    const float* resid;
    if (dim != dd){
      dim3 g((dd+63)/64, (rows+127)/128);
      k_gemm_wmma<<<g, 256, 0, stream>>>(XN, bk.skw, bk.skb, nullptr, SK, rows, dd, dim, 0);
      resid = SK;
    } else {
      resid = xcur;
    }
    if (sq > 1){
      size_t tot = (size_t)B*NqT*dd;
      k_maxpool<<<(unsigned)((tot+255)/256), 256, 0, stream>>>(resid, SKP, B, H, W, dd, sq, OH, OW);
      resid = SKP;
    }

    // x = skip + (o @ proj_w + proj_b)   (WMMA, fused residual)
    const int rq = B*NqT;
    {
      dim3 g((dd+63)/64, (rq+127)/128);
      k_gemm_wmma<<<g, 256, 0, stream>>>(OB, bk.projw, bk.projb, resid, XM, rq, dd, dd, 0);
    }

    // MLP: x = x + fc2(gelu(fc1(LN(x))))
    k_layernorm<<<rq, TB, 0, stream>>>(XM, bk.n2w, bk.n2b, XN, dd);
    const int hid = 4*dd;
    {
      dim3 g((hid+63)/64, (rq+127)/128);
      k_gemm_wmma<<<g, 256, 0, stream>>>(XN, bk.fc1w, bk.fc1b, nullptr, ATT /*hidden*/, rq, hid, dd, 1);
    }
    {
      dim3 g((dd+63)/64, (rq+127)/128);
      k_gemm_wmma<<<g, 256, 0, stream>>>(ATT, bk.fc2w, bk.fc2b, XM, xnext, rq, dd, hid, 0);
    }

    float* tmp = xcur; xcur = xnext; xnext = tmp;
    N = NqT; H = OH; W = OW;
  }

  // final LN(cls) -> head -> softmax
  k_head<<<B, TB, 0, stream>>>(xcur, norm_w, norm_b, head_w, head_b, (float*)d_out, N, 768, 1000);
}
